// TokenContrastiveLearning_43782896616076
// MI455X (gfx1250) — matrix operand order repaired and checked by reading, not compiled, not run on hardware
//
#include <hip/hip_runtime.h>
#include <math.h>

// ---------------------------------------------------------------------------
// TokenContrastiveLearning fused pipeline for gfx1250 (MI455X), wave32 WMMA.
//
//   B=64, L=128, D=768.
//   K1: h = tanh(hidden @ W^T + b) as f16 (fp32 accum via v_wmma_f32_16x16x32_f16)
//       + per-row sum-of-squares (for cosine norms).
//   K2: per (a,b) pair: 128x128 dot tile via WMMA (double-buffered LDS fed by
//       GLOBAL_LOAD_ASYNC_TO_LDS_B128 / s_wait_asynccnt), convert to cosine,
//       masked row-max / col-max (+ first-row/col fallback) -> sim[a,b].
//   K3: loss = -mean(diag(log_softmax(sim/temp))).
//
//   Cross-lane ops use ds_swizzle_b32 immediate patterns (no bpermute addr math).
// ---------------------------------------------------------------------------

typedef __attribute__((ext_vector_type(16))) _Float16 v16h;
typedef __attribute__((ext_vector_type(8)))  float    v8f;

#define DDIM 768
#define TK   32
#define LDH  40   // LDS row stride in f16 (40*2=80B, 16B-aligned, conflict pad)

// ds_swizzle_b32 group-of-32 pattern: src = ((lane & AND) | OR) ^ XOR
#define SWZ_XOR(m)  (((m) << 10) | 0x1f)   // and=0x1f, or=0, xor=m
#define SWZ_AND16   (0x10)                 // src = lane & 16 (half-broadcast lane 0/16)

template <int PAT>
__device__ __forceinline__ float swz(float x) {
    return __builtin_bit_cast(float,
        __builtin_amdgcn_ds_swizzle(__builtin_bit_cast(int, x), PAT));
}

// Single-instruction v_rcp_f32 (plenty of precision for cosine/softmax path).
__device__ __forceinline__ float fast_rcp(float x) {
    return __builtin_amdgcn_rcpf(x);
}

// A fragment: 16x32 f16, per ISA table:
// lanes 0-15: M=lane, VGPR v<4 -> K=2v,2v+1 ; v>=4 -> K=16+2(v-4)..
// lanes 16-31: M=lane-16, K offset +8 within each 16-group.
__device__ __forceinline__ v16h load_frag_A(const _Float16* base, int lane) {
    int hf = (lane >> 4) & 1;
    int lc = lane & 15;
    const _Float16* row = base + lc * LDH;
    v16h a;
#pragma unroll
    for (int v = 0; v < 8; ++v) {
        int k = (v < 4) ? (8 * hf + 2 * v) : (16 + 8 * hf + 2 * (v - 4));
        a[2 * v]     = row[k];
        a[2 * v + 1] = row[k + 1];
    }
    return a;
}

// B fragment: 32x16 f16, B[k,n]; LDS holds B^T rows ([n][k] contiguous).
// lanes 0-15: N=lane, V0..3 -> K=0..7, V4..7 -> K=16..23 ; lanes 16-31: N=lane-16, K offset +8.
__device__ __forceinline__ v16h load_frag_B(const _Float16* base, int lane) {
    int hf = (lane >> 4) & 1;
    int lc = lane & 15;
    const _Float16* row = base + lc * LDH;
    v16h b;
#pragma unroll
    for (int v = 0; v < 8; ++v) {
        int k = (v < 4) ? (8 * hf + 2 * v) : (16 + 8 * hf + 2 * (v - 4));
        b[2 * v]     = row[k];
        b[2 * v + 1] = row[k + 1];
    }
    return b;
}

// Branchless tanh: tanh(x) = sign(x) * (1 - e^{-2|x|}) / (1 + e^{-2|x|}).
// Uses v_exp_f32 + v_rcp_f32 only (no IEEE-div slow sequence).
__device__ __forceinline__ float fast_tanh(float x) {
    float t = __expf(-2.0f * fabsf(x));
    float y = (1.0f - t) * fast_rcp(1.0f + t);
    return __builtin_copysignf(y, x);
}

__global__ void init_kernel(float* __restrict__ nsq) {
    int i = blockIdx.x * blockDim.x + threadIdx.x;
    if (i < 16384) nsq[i] = 0.0f;
}

// ------------------------- K1: projection GEMM -----------------------------
// grid = (12, 256), block = 256.  Tile 64x64, K-step 32.
__global__ __launch_bounds__(256) void gemm_tanh_kernel(
    const float* __restrict__ hidden1, const float* __restrict__ hidden2,
    const float* __restrict__ W, const float* __restrict__ bias,
    _Float16* __restrict__ hout, float* __restrict__ nsq)
{
    __shared__ _Float16 As[64 * LDH];
    __shared__ _Float16 Bs[64 * LDH];

    const int tid  = threadIdx.x;
    const int lane = tid & 31;
    const int w    = tid >> 5;
    const int n0   = blockIdx.x * 64;
    const int m0   = blockIdx.y * 64;

    const float* src = (m0 < 8192) ? (hidden1 + (size_t)m0 * DDIM)
                                   : (hidden2 + (size_t)(m0 - 8192) * DDIM);

    v8f c0 = {}; v8f c1 = {};
    const int fr     = w >> 1;          // 0..3 (16-row strip)
    const int fcbase = (w & 1) * 32;    // col base within 64-wide tile

    const int lrow = tid >> 2;          // 0..63
    const int lcol = (tid & 3) * 8;     // 0,8,16,24

    for (int k0 = 0; k0 < DDIM; k0 += TK) {
        const float* ag = src + (size_t)lrow * DDIM + k0 + lcol;
        const float* bg = W + (size_t)(n0 + lrow) * DDIM + k0 + lcol;
        float4 a0 = ((const float4*)ag)[0];
        float4 a1 = ((const float4*)ag)[1];
        float4 b0 = ((const float4*)bg)[0];
        float4 b1 = ((const float4*)bg)[1];
        _Float16* as = As + lrow * LDH + lcol;
        _Float16* bs = Bs + lrow * LDH + lcol;
        as[0]=(_Float16)a0.x; as[1]=(_Float16)a0.y; as[2]=(_Float16)a0.z; as[3]=(_Float16)a0.w;
        as[4]=(_Float16)a1.x; as[5]=(_Float16)a1.y; as[6]=(_Float16)a1.z; as[7]=(_Float16)a1.w;
        bs[0]=(_Float16)b0.x; bs[1]=(_Float16)b0.y; bs[2]=(_Float16)b0.z; bs[3]=(_Float16)b0.w;
        bs[4]=(_Float16)b1.x; bs[5]=(_Float16)b1.y; bs[6]=(_Float16)b1.z; bs[7]=(_Float16)b1.w;
        __syncthreads();

        v16h af  = load_frag_A(As + (16 * fr) * LDH, lane);
        v16h bf0 = load_frag_B(Bs + fcbase * LDH, lane);
        v16h bf1 = load_frag_B(Bs + (fcbase + 16) * LDH, lane);
        c0 = __builtin_amdgcn_wmma_f32_16x16x32_f16(false, af, false, bf0, (short)0, c0, false, false);
        c1 = __builtin_amdgcn_wmma_f32_16x16x32_f16(false, af, false, bf1, (short)0, c1, false, false);
        __syncthreads();
    }

    const int hf = lane >> 4;
    const int lc = lane & 15;
    const int col0 = n0 + fcbase + lc;
    const int col1 = col0 + 16;
    const float bias0 = bias[col0];
    const float bias1 = bias[col1];
#pragma unroll
    for (int r = 0; r < 8; ++r) {
        int row = m0 + 16 * fr + r + 8 * hf;
        float v0 = fast_tanh(c0[r] + bias0);
        float v1 = fast_tanh(c1[r] + bias1);
        hout[(size_t)row * DDIM + col0] = (_Float16)v0;
        hout[(size_t)row * DDIM + col1] = (_Float16)v1;
        float ss = v0 * v0 + v1 * v1;
        ss += swz<SWZ_XOR(1)>(ss);
        ss += swz<SWZ_XOR(2)>(ss);
        ss += swz<SWZ_XOR(4)>(ss);
        ss += swz<SWZ_XOR(8)>(ss);
        if (lc == 0) atomicAdd(&nsq[row], ss);
    }
}

// ------------------- K2: fused pairwise cosine + RWMD ----------------------
// grid = (64, 64), block = 256.  Each block: one (a,b) pair, full 128x128.
// LDS staging via async global->LDS copies (ASYNCcnt), double-buffered.
__global__ __launch_bounds__(256) void pair_kernel(
    const _Float16* __restrict__ h, const float* __restrict__ nsq,
    const int* __restrict__ mask1, const int* __restrict__ mask2,
    float* __restrict__ sim)
{
    __shared__ _Float16 As[2][128 * LDH];
    __shared__ _Float16 Bs[2][128 * LDH];
    __shared__ float colbuf[8][128];
    __shared__ float row0buf[128];
    __shared__ float norm1[128], norm2[128];
    __shared__ int   m1s[128], m2s[128];
    __shared__ float s_acc[2];

    const int a    = blockIdx.y;
    const int bidx = blockIdx.x;
    const int tid  = threadIdx.x;
    const int lane = tid & 31;
    const int w    = tid >> 5;

    const _Float16* h1 = h + (size_t)a * 128 * DDIM;
    const _Float16* h2 = h + (size_t)(8192 + bidx * 128) * DDIM;

    if (tid < 128) {
        norm1[tid] = sqrtf(nsq[a * 128 + tid]);
        norm2[tid] = sqrtf(nsq[8192 + bidx * 128 + tid]);
        m1s[tid] = mask1[a * 128 + tid];
        m2s[tid] = mask2[bidx * 128 + tid];
    }
    if (tid == 0) { s_acc[0] = 0.0f; s_acc[1] = 0.0f; }

    // Each thread owns 2 16B chunks per tile per K-step.
    const int ch0_row = tid >> 2;                // 0..63   (chunk tid)
    const int ch0_off = (tid & 3) * 8;           // f16 units
    const int ch1_row = (tid + 256) >> 2;        // 64..127 (chunk tid+256)
    const int ch1_off = ch0_off;

    v8f c[8];
#pragma unroll
    for (int n = 0; n < 8; ++n) c[n] = (v8f){};

#define STAGE_ASYNC(k0, buf)                                                   \
    do {                                                                       \
        unsigned da0 = (unsigned)(uintptr_t)&As[(buf)][ch0_row * LDH + ch0_off]; \
        unsigned db0 = (unsigned)(uintptr_t)&Bs[(buf)][ch0_row * LDH + ch0_off]; \
        unsigned da1 = (unsigned)(uintptr_t)&As[(buf)][ch1_row * LDH + ch1_off]; \
        unsigned db1 = (unsigned)(uintptr_t)&Bs[(buf)][ch1_row * LDH + ch1_off]; \
        const _Float16* ga0 = &h1[(size_t)ch0_row * DDIM + (k0) + ch0_off];    \
        const _Float16* gb0 = &h2[(size_t)ch0_row * DDIM + (k0) + ch0_off];    \
        const _Float16* ga1 = &h1[(size_t)ch1_row * DDIM + (k0) + ch1_off];    \
        const _Float16* gb1 = &h2[(size_t)ch1_row * DDIM + (k0) + ch1_off];    \
        asm volatile("global_load_async_to_lds_b128 %0, %1, off"               \
                     :: "v"(da0), "v"(ga0) : "memory");                        \
        asm volatile("global_load_async_to_lds_b128 %0, %1, off"               \
                     :: "v"(db0), "v"(gb0) : "memory");                        \
        asm volatile("global_load_async_to_lds_b128 %0, %1, off"               \
                     :: "v"(da1), "v"(ga1) : "memory");                        \
        asm volatile("global_load_async_to_lds_b128 %0, %1, off"               \
                     :: "v"(db1), "v"(gb1) : "memory");                        \
    } while (0)

    STAGE_ASYNC(0, 0);
    for (int kk = 0; kk < DDIM / TK; ++kk) {
        const int buf = kk & 1;
        asm volatile("s_wait_asynccnt 0x0" ::: "memory");
        __syncthreads();                        // tile `buf` ready for everyone
        if (kk + 1 < DDIM / TK) STAGE_ASYNC((kk + 1) * TK, buf ^ 1);

        v16h af = load_frag_A(&As[buf][(16 * w) * LDH], lane);
#pragma unroll
        for (int n = 0; n < 8; ++n) {
            v16h bf = load_frag_B(&Bs[buf][(16 * n) * LDH], lane);
            c[n] = __builtin_amdgcn_wmma_f32_16x16x32_f16(false, af, false, bf, (short)0, c[n], false, false);
        }
        __syncthreads();                        // all reads of `buf` done
    }
#undef STAGE_ASYNC

    const int hf = lane >> 4;
    const int lc = lane & 15;
    const float EPSC = 1e-8f;

    // Convert dots -> cosine in place (v_rcp_f32, no IEEE-div sequence).
#pragma unroll
    for (int n = 0; n < 8; ++n) {
        int col = 16 * n + lc;
        float n2c = norm2[col];
#pragma unroll
        for (int r = 0; r < 8; ++r) {
            int row = 16 * w + r + 8 * hf;
            float den = fmaxf(norm1[row] * n2c, EPSC);
            c[n][r] = c[n][r] * fast_rcp(den);
        }
    }

    // Row pass: masked max over l2 (or cos[l1,0] if row invalid).
    float s1sum = 0.0f;
#pragma unroll
    for (int r = 0; r < 8; ++r) {
        int row = 16 * w + r + 8 * hf;
        float rowm = -INFINITY;
#pragma unroll
        for (int n = 0; n < 8; ++n) {
            int col = 16 * n + lc;
            float cv = c[n][r];
            if (m2s[col]) rowm = fmaxf(rowm, cv);
        }
        rowm = fmaxf(rowm, swz<SWZ_XOR(1)>(rowm));
        rowm = fmaxf(rowm, swz<SWZ_XOR(2)>(rowm));
        rowm = fmaxf(rowm, swz<SWZ_XOR(4)>(rowm));
        rowm = fmaxf(rowm, swz<SWZ_XOR(8)>(rowm));
        float cos0 = swz<SWZ_AND16>(c[0][r]);   // cos at col 0 of this row-half
        float rowval = m1s[row] ? rowm : cos0;
        s1sum += rowval;
    }
    if (lc == 0) atomicAdd(&s_acc[0], s1sum);

    // Col pass: masked max over l1 (per-wave partials -> LDS).
#pragma unroll
    for (int n = 0; n < 8; ++n) {
        int col = 16 * n + lc;
        float cm = -INFINITY;
#pragma unroll
        for (int r = 0; r < 8; ++r) {
            int row = 16 * w + r + 8 * hf;
            if (m1s[row]) cm = fmaxf(cm, c[n][r]);
        }
        cm = fmaxf(cm, swz<SWZ_XOR(16)>(cm));   // combine halves (rows r and r+8)
        if (hf == 0) colbuf[w][col] = cm;
    }
    if (w == 0 && hf == 0) {
#pragma unroll
        for (int n = 0; n < 8; ++n) row0buf[16 * n + lc] = c[n][0]; // cos[0, col]
    }
    __syncthreads();

    if (tid < 128) {
        float cm = colbuf[0][tid];
#pragma unroll
        for (int i = 1; i < 8; ++i) cm = fmaxf(cm, colbuf[i][tid]);
        float colval = m2s[tid] ? cm : row0buf[tid];
        atomicAdd(&s_acc[1], colval);
    }
    __syncthreads();
    if (tid == 0) sim[a * 64 + bidx] = (s_acc[0] + s_acc[1]) * (1.0f / 256.0f);
}

// ----------------------- K3: contrastive loss ------------------------------
__global__ void loss_kernel(const float* __restrict__ sim, float* __restrict__ out) {
    __shared__ float red[64];
    const int i = threadIdx.x;           // 0..63 rows
    const float invT = 20.0f;            // 1/0.05
    float m = -INFINITY;
    for (int j = 0; j < 64; ++j) m = fmaxf(m, sim[i * 64 + j] * invT);
    float s = 0.0f;
    for (int j = 0; j < 64; ++j) s += __expf(sim[i * 64 + j] * invT - m);
    float lse = m + __logf(s);
    red[i] = lse - sim[i * 64 + i] * invT;
    __syncthreads();
    if (i == 0) {
        float t = 0.0f;
        for (int j = 0; j < 64; ++j) t += red[j];
        out[0] = t * (1.0f / 64.0f);
    }
}

// ---------------------------------------------------------------------------
extern "C" void kernel_launch(void* const* d_in, const int* in_sizes, int n_in,
                              void* d_out, int out_size, void* d_ws, size_t ws_size,
                              hipStream_t stream) {
    const float* hidden1 = (const float*)d_in[0];
    const float* hidden2 = (const float*)d_in[1];
    const int*   mask1   = (const int*)d_in[2];
    const int*   mask2   = (const int*)d_in[3];
    const float* W       = (const float*)d_in[4];
    const float* bias    = (const float*)d_in[5];
    float*       out     = (float*)d_out;

    char* ws = (char*)d_ws;
    _Float16* hbuf = (_Float16*)ws;                               // 16384*768 f16 = 25165824 B
    float*    nsq  = (float*)(ws + (size_t)16384 * 768 * 2);      // 16384 f32
    float*    sim  = (float*)(ws + (size_t)16384 * 768 * 2 + 16384 * 4); // 64*64 f32

    init_kernel<<<dim3(64), dim3(256), 0, stream>>>(nsq);
    gemm_tanh_kernel<<<dim3(12, 256), dim3(256), 0, stream>>>(hidden1, hidden2, W, bias, hbuf, nsq);
    pair_kernel<<<dim3(64, 64), dim3(256), 0, stream>>>(hbuf, nsq, mask1, mask2, sim);
    loss_kernel<<<dim3(1), dim3(64), 0, stream>>>(sim, out);
}
